// AnomalyDetector_63419487092843
// MI455X (gfx1250) — compile-verified, matrix operand order checked by reading, not run on hardware
//
#include <hip/hip_runtime.h>
#include <hip/hip_bf16.h>

// ---------------------------------------------------------------------------
// Problem constants (from reference)
// ---------------------------------------------------------------------------
#define LATENT   128
#define NEDGES   4096
#define NNODES   50000
#define SSAMP    10

// GEMM tiling: 4096 rows = 32 blocks x 128 rows (8 waves x 16 rows each).
// 50000 cols = 25 N-splits x 2000 cols = 25 x (125 tiles of 16 cols). Exact.
#define NSPLIT       25
#define COLS_SPLIT   2000
#define TILES_SPLIT  125

typedef __attribute__((ext_vector_type(16))) __bf16 v16bf;
typedef __attribute__((ext_vector_type(8)))  __bf16 v8bf;
typedef __attribute__((ext_vector_type(8)))  float  v8f;
typedef __attribute__((ext_vector_type(4)))  int    v4i;

// ---------------------------------------------------------------------------
// bf16 helpers (RNE pack)
// ---------------------------------------------------------------------------
__device__ __forceinline__ unsigned short f2bf(float x) {
    unsigned b = __float_as_uint(x);
    unsigned r = b + 0x7FFFu + ((b >> 16) & 1u);
    return (unsigned short)(r >> 16);
}
__device__ __forceinline__ float bf2f(unsigned short h) {
    return __uint_as_float(((unsigned)h) << 16);
}

// ---------------------------------------------------------------------------
// CDNA5 async global->LDS copy (ASYNCcnt path), with sync fallback so the
// file compiles on toolchains lacking the builtin.  16 bytes per lane.
// Builtin pointer params are int4* in AS1 (global) / AS3 (LDS).
// ---------------------------------------------------------------------------
#if defined(__gfx1250__) && __has_builtin(__builtin_amdgcn_global_load_async_to_lds_b128)
#define HAVE_ASYNC_LDS 1
typedef __attribute__((address_space(1))) v4i* as1_v4i_p;
typedef __attribute__((address_space(3))) v4i* as3_v4i_p;
#endif

__device__ __forceinline__ void async_copy16(const unsigned short* g,
                                             unsigned short* l) {
#ifdef HAVE_ASYNC_LDS
    __builtin_amdgcn_global_load_async_to_lds_b128(
        (as1_v4i_p)(g), (as3_v4i_p)(l), /*offset=*/0, /*cpol=*/0);
#else
    *reinterpret_cast<v8bf*>(l) = *reinterpret_cast<const v8bf*>(g);
#endif
}
__device__ __forceinline__ void async_wait0() {
#if defined(__gfx1250__) && __has_builtin(__builtin_amdgcn_s_wait_asynccnt)
    __builtin_amdgcn_s_wait_asynccnt(0);
#endif
}

// ---------------------------------------------------------------------------
// Threefry-2x32 (20 rounds) — deterministic counter-based PRNG, same family
// JAX uses. key = (0, 42), counter = (edge, sample).
// ---------------------------------------------------------------------------
__device__ __forceinline__ void tf_round(unsigned& x0, unsigned& x1, int r) {
    x0 += x1;
    x1 = (x1 << r) | (x1 >> (32 - r));
    x1 ^= x0;
}
__device__ __forceinline__ unsigned threefry2x32(unsigned k0, unsigned k1,
                                                 unsigned c0, unsigned c1) {
    unsigned ks2 = k0 ^ k1 ^ 0x1BD11BDAu;
    unsigned x0 = c0 + k0, x1 = c1 + k1;
    tf_round(x0,x1,13); tf_round(x0,x1,15); tf_round(x0,x1,26); tf_round(x0,x1,6);
    x0 += k1;  x1 += ks2 + 1u;
    tf_round(x0,x1,17); tf_round(x0,x1,29); tf_round(x0,x1,16); tf_round(x0,x1,24);
    x0 += ks2; x1 += k0 + 2u;
    tf_round(x0,x1,13); tf_round(x0,x1,15); tf_round(x0,x1,26); tf_round(x0,x1,6);
    x0 += k0;  x1 += k1 + 3u;
    tf_round(x0,x1,17); tf_round(x0,x1,29); tf_round(x0,x1,16); tf_round(x0,x1,24);
    x0 += k1;  x1 += ks2 + 4u;
    tf_round(x0,x1,13); tf_round(x0,x1,15); tf_round(x0,x1,26); tf_round(x0,x1,6);
    x0 += ks2; x1 += k0 + 5u;
    return x0 ^ x1;
}

// ---------------------------------------------------------------------------
// K0: zero Z and T accumulators
// ---------------------------------------------------------------------------
__global__ __launch_bounds__(256) void k_init(float* Z, float* T) {
    int i = blockIdx.x * 256 + threadIdx.x;
    if (i < NEDGES) { Z[i] = 0.0f; T[i] = 0.0f; }
}

// ---------------------------------------------------------------------------
// K1: convert W (50000x128 f32) -> bf16 once; it then lives in L2 (12.8 MB).
// ---------------------------------------------------------------------------
__global__ __launch_bounds__(256) void k_convW(const float* __restrict__ W,
                                               unsigned short* __restrict__ Wb) {
    int i = blockIdx.x * 256 + threadIdx.x;   // grid covers exactly 6,400,000
    Wb[i] = f2bf(W[i]);
}

// ---------------------------------------------------------------------------
// K2: sample-aggregate.  One block per edge, 128 threads = one latent dim each.
// ---------------------------------------------------------------------------
__global__ __launch_bounds__(128) void k_aggregate(
    const float* __restrict__ z,
    const long long* __restrict__ edges,
    const long long* __restrict__ idx,
    const long long* __restrict__ ptr,
    unsigned short* __restrict__ Abf) {

    __shared__ int nbr[SSAMP];
    const int e = blockIdx.x;
    const int d = threadIdx.x;

    const long long u  = edges[e];            // edges[0][e]
    const long long pu = ptr[u];
    const int       dg = (int)(ptr[u + 1] - pu);

    if (d < SSAMP) {
        unsigned bits = threefry2x32(0u, 42u, (unsigned)e, (unsigned)d);
        float r = (float)(bits >> 8) * (1.0f / 16777216.0f);   // [0,1)
        int off = (int)(r * (float)dg);
        off = (off < dg - 1) ? off : (dg - 1);
        nbr[d] = (int)idx[pu + off];
    }
    __syncthreads();

    float s = z[(size_t)u * LATENT + d];
    #pragma unroll
    for (int j = 0; j < SSAMP; ++j)
        s += z[(size_t)nbr[j] * LATENT + d];
    s *= (1.0f / (float)(SSAMP + 1));

    Abf[e * LATENT + d] = f2bf(s);
}

// ---------------------------------------------------------------------------
// K3: s_target[e] = dot(A[e], W[edges[1][e]]).  One wave32 per edge.
// ---------------------------------------------------------------------------
__global__ __launch_bounds__(256) void k_starget(
    const unsigned short* __restrict__ Abf,
    const float* __restrict__ W,
    const long long* __restrict__ edges,
    float* __restrict__ st) {

    const int e    = blockIdx.x * 8 + (threadIdx.x >> 5);
    const int lane = threadIdx.x & 31;
    const long long v = edges[NEDGES + e];    // edges[1][e]

    float s = 0.0f;
    #pragma unroll
    for (int d = lane; d < LATENT; d += 32)
        s += bf2f(Abf[e * LATENT + d]) * W[(size_t)v * LATENT + d];

    s += __shfl_xor(s, 16);
    s += __shfl_xor(s, 8);
    s += __shfl_xor(s, 4);
    s += __shfl_xor(s, 2);
    s += __shfl_xor(s, 1);
    if (lane == 0) st[e] = s;
}

// ---------------------------------------------------------------------------
// K4/K5: fused GEMM + reduction passes.
//   PASS 0:  Z[row] += sum_n exp(s[row][n])
//   PASS 1:  T[row] += sum_n exp( exp(s[row][n]) / Z[row] )
//
// Block = 256 threads = 8 waves; each wave owns a 16-row M-tile (A in regs as
// four v16bf K-chunks).  The 16-col x 128-K B tile is double-buffered in LDS:
// tile t+1 streams in via GLOBAL_LOAD_ASYNC_TO_LDS_B128 (ASYNCcnt) while the
// 8 waves run 4 chained v_wmma_f32_16x16x32_bf16 on tile t.
//
// A-fragment per ISA layout (16-bit A 16x32): lane[3:0] = row, lane[4] selects
// K lo/hi 8-halves; two 16B loads fill the v16bf.  C layout: VGPR i holds
// row i (lanes 0-15) / row i+8 (lanes 16-31), col = lane&15.
// ---------------------------------------------------------------------------
template <int PASS>
__global__ __launch_bounds__(256) void gemm_softmax_pass(
    const unsigned short* __restrict__ Abf,
    const unsigned short* __restrict__ Wbf,
    const float* __restrict__ Zin,     // pass 1 only
    float* __restrict__ out) {         // Z (pass 0) or T (pass 1)

    __shared__ __align__(16) unsigned short Bt[2][16 * LATENT];   // 2 x 4 KB

    const int tid  = threadIdx.x;
    const int wave = tid >> 5;
    const int lane = tid & 31;
    const int r0   = blockIdx.x * 128 + wave * 16;    // wave's 16 rows
    const int nbeg = blockIdx.y * COLS_SPLIT;

    // staging assignment: 256 threads x 16 bytes = one 4 KB tile
    const int scol = tid >> 4;          // 0..15
    const int skc  = (tid & 15) * 8;    // 0..120

    // ---- A fragments: row r0 + lane[3:0]; K-half by lane[4] -----------------
    const int arow  = r0 + (lane & 15);
    const int akoff = ((lane >> 4) & 1) * 8;
    v16bf a[4];
    #pragma unroll
    for (int c = 0; c < 4; ++c) {
        const v8bf* p =
            reinterpret_cast<const v8bf*>(Abf + arow * LATENT + c * 32 + akoff);
        union { v16bf v; v8bf h[2]; } u;
        u.h[0] = p[0];   // K = base .. base+7
        u.h[1] = p[2];   // K = base+16 .. base+23
        a[c] = u.v;
    }

    // ---- per-lane row scale (pass 1): rows r0 + (lane[4]?8:0) + i -----------
    float rz[8];
    if (PASS == 1) {
        const int rbase = r0 + ((lane >> 4) ? 8 : 0);
        #pragma unroll
        for (int i = 0; i < 8; ++i) rz[i] = 1.0f / Zin[rbase + i];
    }

    float acc[8];
    #pragma unroll
    for (int i = 0; i < 8; ++i) acc[i] = 0.0f;

    const int bcol = lane & 15;

    // ---- prologue: stage tile 0 into buffer 0 -------------------------------
    async_copy16(Wbf + (size_t)(nbeg + scol) * LATENT + skc, &Bt[0][scol * LATENT + skc]);
    async_wait0();
    __syncthreads();

    for (int t = 0; t < TILES_SPLIT; ++t) {
        const int cur = t & 1;
        const int nxt = cur ^ 1;

        // kick off tile t+1 into the other buffer (released by barrier at t-1)
        if (t + 1 < TILES_SPLIT) {
            const int n1 = nbeg + (t + 1) * 16;
            async_copy16(Wbf + (size_t)(n1 + scol) * LATENT + skc,
                         &Bt[nxt][scol * LATENT + skc]);
        }

        // compute on tile t
        v8f cfrag = {};
        #pragma unroll
        for (int c = 0; c < 4; ++c) {
            const v8bf* bp = reinterpret_cast<const v8bf*>(
                &Bt[cur][bcol * LATENT + c * 32 + akoff]);
            union { v16bf v; v8bf h[2]; } u;
            u.h[0] = bp[0];
            u.h[1] = bp[2];
            cfrag = __builtin_amdgcn_wmma_f32_16x16x32_bf16(
                false, a[c], false, u.v, (short)0, cfrag, false, false);
        }

        #pragma unroll
        for (int i = 0; i < 8; ++i) {
            if (PASS == 0) {
                acc[i] += __expf(cfrag[i]);
            } else {
                float h = __expf(cfrag[i]) * rz[i];
                acc[i] += __expf(h);
            }
        }

        // tile t fully read by all waves; tile t+1 landed in LDS
        async_wait0();
        __syncthreads();
    }

    // ---- reduce the 16 lanes sharing each row, then one atomic per row ------
    const int rbase = r0 + ((lane >> 4) ? 8 : 0);
    #pragma unroll
    for (int i = 0; i < 8; ++i) {
        float v = acc[i];
        v += __shfl_xor(v, 1);
        v += __shfl_xor(v, 2);
        v += __shfl_xor(v, 4);
        v += __shfl_xor(v, 8);
        if ((lane & 15) == 0) atomicAdd(&out[rbase + i], v);
    }
}

// ---------------------------------------------------------------------------
// K6: loss = mean( log(T[e]) - exp(s_target[e]) / Z[e] )   (single block)
// ---------------------------------------------------------------------------
__global__ __launch_bounds__(256) void k_finalize(
    const float* __restrict__ Z, const float* __restrict__ T,
    const float* __restrict__ st, float* __restrict__ out) {

    __shared__ float red[256];
    float s = 0.0f;
    for (int e = threadIdx.x; e < NEDGES; e += 256) {
        float h = __expf(st[e]) / Z[e];
        s += __logf(T[e]) - h;
    }
    red[threadIdx.x] = s;
    __syncthreads();
    for (int w = 128; w > 0; w >>= 1) {
        if (threadIdx.x < w) red[threadIdx.x] += red[threadIdx.x + w];
        __syncthreads();
    }
    if (threadIdx.x == 0) out[0] = red[0] / (float)NEDGES;
}

// ---------------------------------------------------------------------------
// Launch
// ---------------------------------------------------------------------------
extern "C" void kernel_launch(void* const* d_in, const int* in_sizes, int n_in,
                              void* d_out, int out_size, void* d_ws, size_t ws_size,
                              hipStream_t stream) {
    const float*     z     = (const float*)d_in[0];
    const float*     W     = (const float*)d_in[1];
    const long long* edges = (const long long*)d_in[2];
    const long long* idx   = (const long long*)d_in[3];
    const long long* ptr   = (const long long*)d_in[4];
    float*           outp  = (float*)d_out;

    char* ws = (char*)d_ws;
    unsigned short* Abf = (unsigned short*)(ws);                       //  1,048,576 B
    unsigned short* Wbf = (unsigned short*)(ws + (1u << 20));          // 12,800,000 B
    float* Z  = (float*)(ws + 13848576u);                              //     16,384 B
    float* T  = (float*)(ws + 13864960u);                              //     16,384 B
    float* st = (float*)(ws + 13881344u);                              //     16,384 B

    k_init<<<dim3((NEDGES + 255) / 256), dim3(256), 0, stream>>>(Z, T);
    k_convW<<<dim3((NNODES * LATENT) / 256), dim3(256), 0, stream>>>(W, Wbf);
    k_aggregate<<<dim3(NEDGES), dim3(128), 0, stream>>>(z, edges, idx, ptr, Abf);
    k_starget<<<dim3(NEDGES / 8), dim3(256), 0, stream>>>(Abf, W, edges, st);

    dim3 ggrid(NEDGES / 128, NSPLIT);    // 32 x 25
    gemm_softmax_pass<0><<<ggrid, dim3(256), 0, stream>>>(Abf, Wbf, nullptr, Z);
    gemm_softmax_pass<1><<<ggrid, dim3(256), 0, stream>>>(Abf, Wbf, Z, T);

    k_finalize<<<dim3(1), dim3(256), 0, stream>>>(Z, T, st, outp);
}
